// I2MoE_43671227465946
// MI455X (gfx1250) — matrix operand order repaired and checked by reading, not compiled, not run on hardware
//
#include <hip/hip_runtime.h>

// ---------------------------------------------------------------------------
// I2MoE forward for MI455X (gfx1250): bf16 WMMA GEMMs with double-buffered
// async global->LDS staging (ASYNCcnt pipeline), wave32.
// ---------------------------------------------------------------------------

typedef __bf16 bf16;
typedef __bf16 v16bf __attribute__((ext_vector_type(16)));
typedef float  v8f   __attribute__((ext_vector_type(8)));
typedef unsigned int u32x4 __attribute__((ext_vector_type(4)));
union FragV { u32x4 u[2]; v16bf v; };

#define ACT_NONE 0
#define ACT_SILU 1
#define ACT_GELU 2
#define ACT_RELU 3
#define ACT_SOFTPLUS 4

__device__ __forceinline__ float sigm(float x){ return 1.f/(1.f+__expf(-x)); }
__device__ __forceinline__ float siluf(float x){ return x*sigm(x); }
__device__ __forceinline__ float geluf(float x){
  float t = 0.7978845608028654f*(x+0.044715f*x*x*x);
  return 0.5f*x*(1.f+tanhf(t));
}
__device__ __forceinline__ float softplusf(float x){ return x>20.f?x:log1pf(__expf(x)); }
__device__ __forceinline__ float applyAct(float v,int act){
  switch(act){
    case ACT_SILU: return siluf(v);
    case ACT_GELU: return geluf(v);
    case ACT_RELU: return fmaxf(v,0.f);
    case ACT_SOFTPLUS: return softplusf(v);
  }
  return v;
}

// ---- CDNA5 async global->LDS copies (ASYNCcnt path) -----------------------
__device__ __forceinline__ void async_copy_b128(unsigned lds_off, const void* gptr){
  asm volatile("global_load_async_to_lds_b128 %0, %1, off"
               :: "v"(lds_off), "v"(gptr) : "memory");
}
__device__ __forceinline__ void async_wait0(){   // drain all async transfers
  asm volatile("s_wait_asynccnt 0x0" ::: "memory");
}
__device__ __forceinline__ void async_wait3(){   // keep newest tile (3) in flight
  asm volatile("s_wait_asynccnt 0x3" ::: "memory");
}

// ---------------------------------------------------------------------------
// GEMM: C[M,N] = A[M,K] * W[N,K]^T (+bias, +act). A,W bf16; C fp32 and/or bf16.
// Block tile = 64x128, K step 32. 8 wave32 waves as 4x2; each wave computes a
// 16x64 strip via four 16x16x32 bf16 WMMAs per K step. Global->LDS staging is
// double-buffered with global_load_async_to_lds_b128: tile k+1 is issued
// before waiting on tile k (s_wait_asynccnt 3 leaves it in flight).
// ---------------------------------------------------------------------------
__global__ __launch_bounds__(256) void k_gemm(
    const bf16* __restrict__ A, long lda,
    const bf16* __restrict__ W, long ldw,
    const float* __restrict__ bias,
    float* __restrict__ C, bf16* __restrict__ Cb, long ldc,
    int M, int N, int K, int act)
{
  // per buffer: A tile 64x32 (4KB) then B tile 128x32 (8KB); two buffers.
  __shared__ __align__(16) bf16 smem[2 * 6144];
  const int tid  = threadIdx.x;
  const int lane = tid & 31, wave = tid >> 5;
  const int wm = wave >> 1, wn = wave & 1;
  const int bm = blockIdx.y * 64, bn = blockIdx.x * 128;

  v8f acc[4] = {};

  // --- staging maps: A -> 1 x b128 / thread, B -> 2 x b128 / thread --------
  const int rA = tid >> 2, cA = tid & 3;               // A row 0..63, 16B chunk
  int aRow = bm + rA; if (aRow >= M) aRow = M - 1;
  const char* aBase = (const char*)(A + (size_t)aRow * lda) + cA * 16;

  const int i0 = tid * 2, i1 = tid * 2 + 1;            // B transfers
  const int rB0 = i0 >> 2, cB0 = i0 & 3;               // B rows 0..127
  const int rB1 = i1 >> 2, cB1 = i1 & 3;
  int wRow0 = bn + rB0; if (wRow0 >= N) wRow0 = N - 1;
  int wRow1 = bn + rB1; if (wRow1 >= N) wRow1 = N - 1;
  const char* wBase0 = (const char*)(W + (size_t)wRow0 * ldw) + cB0 * 16;
  const char* wBase1 = (const char*)(W + (size_t)wRow1 * ldw) + cB1 * 16;

  const unsigned ldsBase = (unsigned)(unsigned long long)(void*)smem;
  const unsigned aL  = ldsBase + (unsigned)(rA * 64 + cA * 16);
  const unsigned bL0 = ldsBase + 4096u + (unsigned)(rB0 * 64 + cB0 * 16);
  const unsigned bL1 = ldsBase + 4096u + (unsigned)(rB1 * 64 + cB1 * 16);

  auto issue = [&](int t) {
    unsigned off = (t & 1) ? 12288u : 0u;              // buffer select (bytes)
    size_t go = (size_t)t * 64;                        // 32 bf16 K-step
    async_copy_b128(aL  + off, aBase  + go);
    async_copy_b128(bL0 + off, wBase0 + go);
    async_copy_b128(bL1 + off, wBase1 + go);
  };

  // --- fragment addressing (documented wave32 layouts) ---------------------
  const int arow = wm * 16 + (lane & 15);
  const int kA   = (lane >> 4);         // A: lanes 0-15 K0..7/16..23, 16-31 K8..15/24..31
  const int nb   = wn * 64 + (lane & 15);
  const int kB   = (lane >> 4);         // B: lanes 0-15 K0..7/16..23, 16-31 K8..15/24..31

  const int kIters = K >> 5;
  issue(0);
  for (int kk = 0; kk < kIters; ++kk) {
    if (kk + 1 < kIters) { issue(kk + 1); async_wait3(); }
    else                 { async_wait0(); }
    __syncthreads();                                   // tile kk visible

    const bf16* buf = smem + (kk & 1) * 6144;
    const u32x4* SA = (const u32x4*)buf;
    const u32x4* SB = (const u32x4*)(buf + 2048);

    FragV af;
    af.u[0] = SA[arow * 4 + kA];
    af.u[1] = SA[arow * 4 + kA + 2];
#pragma unroll
    for (int s = 0; s < 4; ++s) {
      FragV bf_;
      int nbs = nb + s * 16;
      bf_.u[0] = SB[nbs * 4 + kB];
      bf_.u[1] = SB[nbs * 4 + kB + 2];
      acc[s] = __builtin_amdgcn_wmma_f32_16x16x32_bf16(false, af.v, false, bf_.v,
                                                       (short)0, acc[s], false, false);
    }
    __syncthreads();                                   // done reading buf before reuse
  }

  // --- epilogue: D layout -> VGPR rr: rows rr (lanes 0-15) / rr+8 (16-31) --
  const int mBase = bm + wm * 16 + 8 * (lane >> 4);
  const int nBase = bn + wn * 64 + (lane & 15);
#pragma unroll
  for (int s = 0; s < 4; ++s) {
    int n = nBase + s * 16;
    if (n >= N) continue;
    float bv = bias ? bias[n] : 0.f;
#pragma unroll
    for (int rr = 0; rr < 8; ++rr) {
      int m = mBase + rr;
      if (m >= M) continue;
      float v = applyAct(acc[s][rr] + bv, act);
      if (C)  C [(size_t)m * ldc + n] = v;
      if (Cb) Cb[(size_t)m * ldc + n] = (bf16)v;
    }
  }
}

// ---------------------------------------------------------------------------
// Elementwise / helper kernels
// ---------------------------------------------------------------------------
__global__ void k_cvt(const float* __restrict__ s, bf16* __restrict__ d, long n){
  long i = (long)blockIdx.x * 256 + threadIdx.x;
  if (i < n) d[i] = (bf16)s[i];
}
// s[K,N] (row-major) -> d[N,K]
__global__ void k_cvtT(const float* __restrict__ s, bf16* __restrict__ d, int K, int N){
  int n = blockIdx.x * 16 + threadIdx.x;
  int k = blockIdx.y * 16 + threadIdx.y;
  if (n < N && k < K) d[(size_t)n * K + k] = (bf16)s[(size_t)k * N + n];
}
// slice x[:, s:s+L, :] (Lpitch tokens per batch), optional flip along L
__global__ void k_slice(const float* __restrict__ x, int Lpitch, int sOff, int L,
                        int flip, float* __restrict__ of, bf16* __restrict__ ob){
  long idx = (long)blockIdx.x * 256 + threadIdx.x;
  long tot = (long)32 * L * 1024;
  if (idx >= tot) return;
  int d = idx & 1023;
  long row = idx >> 10;
  int b = row / L, l = row % L;
  int ls = flip ? (L - 1 - l) : l;
  float v = x[((size_t)b * Lpitch + sOff + ls) * 1024 + d];
  if (of) of[idx] = v;
  if (ob) ob[idx] = (bf16)v;
}
// causal depthwise conv + SiLU. xi strided (stride xiS), weights [di,K]
__global__ void k_conv(const float* __restrict__ xi, int xiS,
                       const float* __restrict__ cw, const float* __restrict__ cb,
                       int L, int di, int K,
                       float* __restrict__ xcF, bf16* __restrict__ xcB){
  long idx = (long)blockIdx.x * 256 + threadIdx.x;
  long tot = (long)32 * L * di;
  if (idx >= tot) return;
  int d = idx % di;
  long row = idx / di;
  int b = row / L, l = row % L;
  float s = cb[d];
  for (int k = 0; k < K; ++k) {
    int lp = l - (K - 1) + k;
    if (lp >= 0) s += xi[((size_t)(b * L + lp)) * xiS + d] * cw[(size_t)d * K + k];
  }
  s = siluf(s);
  xcF[(size_t)row * di + d] = s;
  xcB[(size_t)row * di + d] = (bf16)s;
}
// selective scan: h = exp(dl*A)*h + dl*x*B ; y = h.C + x*D ; y *= silu(z)
__global__ void k_scan(const float* __restrict__ delta, const float* __restrict__ xc,
                       const float* __restrict__ xpj, int nxp, int dtr, int ds,
                       const float* __restrict__ z, int zS,
                       const float* __restrict__ Alog, const float* __restrict__ Dv,
                       bf16* __restrict__ yB, int L, int di){
  int d = blockIdx.x * 256 + threadIdx.x;
  int b = blockIdx.y;
  __shared__ float sB[16], sC[16];
  float a[16], h[16];
  for (int s = 0; s < ds; ++s) { a[s] = -__expf(Alog[(size_t)d * ds + s]); h[s] = 0.f; }
  float Dd = Dv[d];
  for (int l = 0; l < L; ++l) {
    size_t row = (size_t)b * L + l;
    __syncthreads();
    if (threadIdx.x < (unsigned)(2 * ds))
      ((threadIdx.x < (unsigned)ds) ? sB[threadIdx.x] : sC[threadIdx.x - ds]) =
          xpj[row * nxp + dtr + threadIdx.x];
    __syncthreads();
    float dl = delta[row * di + d];
    float xv = xc[row * di + d];
    float dx = dl * xv, y = 0.f;
    for (int s = 0; s < ds; ++s) {
      h[s] = __expf(dl * a[s]) * h[s] + dx * sB[s];
      y += h[s] * sC[s];
    }
    y += xv * Dd;
    float zv = z[row * zS + d];
    y *= zv * sigm(zv);
    yB[row * di + d] = (bf16)y;
  }
}
// LayerNorm over dim of (xres + a + flip(b)); optional post-GELU; strided dst
__global__ void k_ln(const float* __restrict__ xres, const float* __restrict__ a,
                     const float* __restrict__ bsrc, int flipB,
                     const float* __restrict__ g, const float* __restrict__ beta,
                     int L, int dim, int postGelu,
                     float* __restrict__ outF, bf16* __restrict__ outB,
                     int dstLpitch, int dstOff){
  int row = blockIdx.x;
  int b = row / L, l = row % L;
  size_t so  = (size_t)row * dim;
  size_t sob = ((size_t)b * L + (flipB ? (L - 1 - l) : l)) * dim;
  float vloc[4]; int cnt = 0;
  float p1 = 0.f, p2 = 0.f;
  for (int i = threadIdx.x; i < dim; i += 256) {
    float v = 0.f;
    if (xres) v += xres[so + i];
    if (a)    v += a[so + i];
    if (bsrc) v += bsrc[sob + i];
    vloc[cnt++] = v; p1 += v; p2 += v * v;
  }
  __shared__ float sh1[8], sh2[8];
  int lane = threadIdx.x & 31, wave = threadIdx.x >> 5;
  for (int m = 16; m; m >>= 1) { p1 += __shfl_xor(p1, m, 32); p2 += __shfl_xor(p2, m, 32); }
  if (lane == 0) { sh1[wave] = p1; sh2[wave] = p2; }
  __syncthreads();
  float t1 = 0.f, t2 = 0.f;
  for (int w = 0; w < 8; ++w) { t1 += sh1[w]; t2 += sh2[w]; }
  float mu = t1 / dim;
  float var = t2 / dim - mu * mu;
  float rstd = rsqrtf(var + 1e-5f);
  size_t drow = ((size_t)b * dstLpitch + dstOff + l) * dim;
  cnt = 0;
  for (int i = threadIdx.x; i < dim; i += 256) {
    float v = (vloc[cnt++] - mu) * rstd * g[i] + beta[i];
    if (postGelu) v = geluf(v);
    if (outF) outF[drow + i] = v;
    if (outB) outB[drow + i] = (bf16)v;
  }
}
// mean over L tokens -> feats[b, slot, :]
__global__ void k_mean(const float* __restrict__ src, int Lpitch, int L,
                       float* __restrict__ dF, bf16* __restrict__ dB, int slot){
  int idx = blockIdx.x * 256 + threadIdx.x;
  if (idx >= 32 * 1024) return;
  int b = idx >> 10, d = idx & 1023;
  float s = 0.f;
  for (int l = 0; l < L; ++l) s += src[((size_t)b * Lpitch + l) * 1024 + d];
  s /= (float)L;
  size_t o = ((size_t)b * 4 + slot) * 1024 + d;
  dF[o] = s; dB[o] = (bf16)s;
}
// rowwise l2-normalize (optionally / (softplus(tau)+1e-6)) -> bf16
__global__ void k_l2scale(const float* __restrict__ src, bf16* __restrict__ dst,
                          int dim, const float* __restrict__ tau){
  int row = blockIdx.x;
  size_t o = (size_t)row * dim;
  float p = 0.f;
  for (int i = threadIdx.x; i < dim; i += 256) { float v = src[o + i]; p += v * v; }
  __shared__ float sh[8];
  int lane = threadIdx.x & 31, wave = threadIdx.x >> 5;
  for (int m = 16; m; m >>= 1) p += __shfl_xor(p, m, 32);
  if (lane == 0) sh[wave] = p;
  __syncthreads();
  float t = 0.f;
  for (int w = 0; w < 8; ++w) t += sh[w];
  float den = fmaxf(sqrtf(t), 1e-12f);
  if (tau) den *= (softplusf(tau[0]) + 1e-6f);
  float inv = 1.f / den;
  for (int i = threadIdx.x; i < dim; i += 256) dst[o + i] = (bf16)(src[o + i] * inv);
}
// MHA core (nh=8, hd=128, Lk=4): one block per (b,q); wave per head
__global__ void k_attn(const float* __restrict__ qp, const float* __restrict__ kp,
                       const float* __restrict__ vp, int Lq,
                       float* __restrict__ o, bf16* __restrict__ ob){
  int bq = blockIdx.x;
  int b = bq / Lq;
  int wave = threadIdx.x >> 5, lane = threadIdx.x & 31;
  const float* qr = qp + (size_t)bq * 1024 + wave * 128;
  float qv[4];
  for (int j = 0; j < 4; ++j) qv[j] = qr[lane * 4 + j];
  float sc[4];
  for (int kk = 0; kk < 4; ++kk) {
    const float* kr = kp + ((size_t)b * 4 + kk) * 1024 + wave * 128;
    float s = 0.f;
    for (int j = 0; j < 4; ++j) s += qv[j] * kr[lane * 4 + j];
    for (int m = 16; m; m >>= 1) s += __shfl_xor(s, m, 32);
    sc[kk] = s * 0.08838834764831845f;  // 1/sqrt(128)
  }
  float mx = fmaxf(fmaxf(sc[0], sc[1]), fmaxf(sc[2], sc[3]));
  float e[4], sum = 0.f;
  for (int kk = 0; kk < 4; ++kk) { e[kk] = __expf(sc[kk] - mx); sum += e[kk]; }
  float inv = 1.f / sum;
  float acc[4] = {0.f, 0.f, 0.f, 0.f};
  for (int kk = 0; kk < 4; ++kk) {
    const float* vr = vp + ((size_t)b * 4 + kk) * 1024 + wave * 128;
    float p = e[kk] * inv;
    for (int j = 0; j < 4; ++j) acc[j] += p * vr[lane * 4 + j];
  }
  size_t base = (size_t)bq * 1024 + wave * 128 + lane * 4;
  for (int j = 0; j < 4; ++j) { o[base + j] = acc[j]; ob[base + j] = (bf16)acc[j]; }
}
// out[row] = dot(h[row,:dim], w) + b0   (wave per row)
__global__ void k_rowdot(const float* __restrict__ h, const float* __restrict__ w,
                         const float* __restrict__ b0, float* __restrict__ out,
                         int rows, int dim){
  int row = blockIdx.x * 8 + (threadIdx.x >> 5);
  int lane = threadIdx.x & 31;
  if (row >= rows) return;
  float s = 0.f;
  for (int i = lane; i < dim; i += 32) s += h[(size_t)row * dim + i] * w[i];
  for (int m = 16; m; m >>= 1) s += __shfl_xor(s, m, 32);
  if (lane == 0) out[row] = s + b0[0];
}
__global__ void k_softmax(const float* __restrict__ in, float* __restrict__ out,
                          int rows, int cols, const float* __restrict__ tau){
  int r = blockIdx.x * 64 + threadIdx.x;
  if (r >= rows) return;
  float t = tau ? tau[0] : 1.f;
  float v[8], mx = -1e30f;
  for (int c = 0; c < cols; ++c) { v[c] = in[(size_t)r * cols + c] / t; mx = fmaxf(mx, v[c]); }
  float sum = 0.f;
  for (int c = 0; c < cols; ++c) { v[c] = __expf(v[c] - mx); sum += v[c]; }
  float inv = 1.f / sum;
  for (int c = 0; c < cols; ++c) out[(size_t)r * cols + c] = v[c] * inv;
}
__global__ void k_combine(const float* __restrict__ tokens, const float* __restrict__ syn,
                          const float* __restrict__ red, const float* __restrict__ modw,
                          const float* __restrict__ interw,
                          float* __restrict__ out, float* __restrict__ comb){
  long idx = (long)blockIdx.x * 256 + threadIdx.x;
  const long tot = (long)32 * 39 * 1024;
  if (idx < tot) {
    int b = idx / (39 * 1024);
    int l = (idx / 1024) % 39;
    int m = (l < 1) ? 0 : (l < 13) ? 1 : (l < 20) ? 2 : 3;
    const float* iw = interw + (size_t)b * 3;
    out[idx] = iw[0] * modw[(size_t)b * 4 + m] * tokens[idx]
             + iw[1] * syn[idx] + iw[2] * red[idx];
  }
  if (idx < 32 * 7) {
    int b = idx / 7, c = idx % 7;
    comb[idx] = (c < 4) ? modw[(size_t)b * 4 + c] : interw[(size_t)b * 3 + (c - 4)];
  }
}

// ---------------------------------------------------------------------------
// Host orchestration
// ---------------------------------------------------------------------------
struct Wspace { char* p; size_t off, cap; };
static void* walloc(Wspace& w, size_t bytes){
  size_t a = (w.off + 255) & ~(size_t)255;
  w.off = a + bytes;
  return w.p + a;
}

static void gemm(hipStream_t st, const bf16* A, long lda, const bf16* W, long ldw,
                 const float* bias, float* C, bf16* Cb, long ldc,
                 int M, int N, int K, int act){
  dim3 g((N + 127) / 128, (M + 63) / 64);
  k_gemm<<<g, 256, 0, st>>>(A, lda, W, ldw, bias, C, Cb, ldc, M, N, K, act);
}

struct Cfg { int di, dconv, ds, dff; };
static const Cfg CFGS[6] = {
  {1024, 2, 16, 1024},   // fire
  {2048, 4, 16, 4096},   // weather
  {2048, 4, 16, 2048},   // terrain
  {2048, 4, 16, 3072},   // modis
  {3072, 4, 16, 6144},   // synergy
  {2048, 2, 8,  3072},   // redundancy
};
static const int RSTART[4] = {0, 1, 13, 20};
static const int RLEN[4]   = {1, 12, 7, 19};

struct MambaW {
  const bf16 *inproj, *xproj, *dtproj, *outproj;
  const float *convw, *convb, *dtb, *Alog, *Dv;
};
struct EncW {
  MambaW f, r;
  const float *ln1g, *ln1b, *ln2g, *ln2b, *b1, *b2;
  const bf16 *w1t, *w2t;
};

static void run_encoder(hipStream_t st, Wspace& ws, const float* src, int Lpitch,
                        int sOff, int L, const Cfg& cfg, const EncW& W,
                        float* dstF, bf16* dstB, int dstLpitch, int dstOff){
  size_t mark = ws.off;
  const int D = 1024, M = 32 * L, di = cfg.di, nxp = 64 + 2 * cfg.ds;
  float* Xf  = (float*)walloc(ws, (size_t)M * D * 4);
  bf16*  Xb  = (bf16*) walloc(ws, (size_t)M * D * 2);
  bf16*  Xrb = (bf16*) walloc(ws, (size_t)M * D * 2);
  long tot = (long)M * D;
  k_slice<<<(tot + 255) / 256, 256, 0, st>>>(src, Lpitch, sOff, L, 0, Xf, Xb);
  k_slice<<<(tot + 255) / 256, 256, 0, st>>>(src, Lpitch, sOff, L, 1, nullptr, Xrb);

  float* outF_ = (float*)walloc(ws, (size_t)M * D * 4);
  float* outR_ = (float*)walloc(ws, (size_t)M * D * 4);
  size_t mark2 = ws.off;
  for (int dir = 0; dir < 2; ++dir) {
    ws.off = mark2;
    const MambaW& mw = dir ? W.r : W.f;
    const bf16* Ain = dir ? Xrb : Xb;
    float* outDir = dir ? outR_ : outF_;

    float* inp = (float*)walloc(ws, (size_t)M * 2 * di * 4);
    gemm(st, Ain, D, mw.inproj, D, nullptr, inp, nullptr, 2 * di, M, 2 * di, D, ACT_NONE);

    float* xcF = (float*)walloc(ws, (size_t)M * di * 4);
    bf16*  xcB = (bf16*) walloc(ws, (size_t)M * di * 2);
    long ct = (long)M * di;
    k_conv<<<(ct + 255) / 256, 256, 0, st>>>(inp, 2 * di, mw.convw, mw.convb,
                                             L, di, cfg.dconv, xcF, xcB);

    float* xpF = (float*)walloc(ws, (size_t)M * nxp * 4);
    bf16*  xpB = (bf16*) walloc(ws, (size_t)M * nxp * 2);
    gemm(st, xcB, di, mw.xproj, di, nullptr, xpF, xpB, nxp, M, nxp, di, ACT_NONE);

    float* delta = (float*)walloc(ws, (size_t)M * di * 4);
    gemm(st, xpB, nxp, mw.dtproj, 64, mw.dtb, delta, nullptr, di, M, di, 64, ACT_SOFTPLUS);

    bf16* yB = (bf16*)walloc(ws, (size_t)M * di * 2);
    k_scan<<<dim3(di / 256, 32), 256, 0, st>>>(delta, xcF, xpF, nxp, 64, cfg.ds,
                                               inp + di, 2 * di, mw.Alog, mw.Dv,
                                               yB, L, di);
    gemm(st, yB, di, mw.outproj, di, nullptr, outDir, nullptr, D, M, D, di, ACT_NONE);
  }
  ws.off = mark2;

  float* x1F = (float*)walloc(ws, (size_t)M * D * 4);
  bf16*  x1B = (bf16*) walloc(ws, (size_t)M * D * 2);
  k_ln<<<M, 256, 0, st>>>(Xf, outF_, outR_, 1, W.ln1g, W.ln1b, L, D, 0,
                          x1F, x1B, L, 0);

  bf16* hB = (bf16*)walloc(ws, (size_t)M * cfg.dff * 2);
  gemm(st, x1B, D, W.w1t, D, W.b1, nullptr, hB, cfg.dff, M, cfg.dff, D, ACT_GELU);
  float* y2 = (float*)walloc(ws, (size_t)M * D * 4);
  gemm(st, hB, cfg.dff, W.w2t, cfg.dff, W.b2, y2, nullptr, D, M, D, cfg.dff, ACT_NONE);

  k_ln<<<M, 256, 0, st>>>(x1F, y2, nullptr, 0, W.ln2g, W.ln2b, L, D, 0,
                          dstF, dstB, dstLpitch, dstOff);
  ws.off = mark;
}

static void run_mha(hipStream_t st, Wspace& ws, const bf16* qA, int Mq, int Lq,
                    const bf16* kA, const bf16* vA,
                    const bf16* inW, const float* inB,
                    const bf16* outW, const float* outB_,
                    float* oF, bf16* oB){
  size_t mark = ws.off;
  const int D = 1024, Mk = 128;
  float* qp = (float*)walloc(ws, (size_t)Mq * D * 4);
  float* kp = (float*)walloc(ws, (size_t)Mk * D * 4);
  float* vp = (float*)walloc(ws, (size_t)Mk * D * 4);
  gemm(st, qA, D, inW,                     D, inB,         qp, nullptr, D, Mq, D, D, ACT_NONE);
  gemm(st, kA, D, inW + (size_t)D * D,     D, inB + D,     kp, nullptr, D, Mk, D, D, ACT_NONE);
  gemm(st, vA, D, inW + (size_t)2 * D * D, D, inB + 2 * D, vp, nullptr, D, Mk, D, D, ACT_NONE);
  float* at  = (float*)walloc(ws, (size_t)Mq * D * 4);
  bf16*  atB = (bf16*) walloc(ws, (size_t)Mq * D * 2);
  k_attn<<<Mq, 256, 0, st>>>(qp, kp, vp, Lq, at, atB);
  gemm(st, atB, D, outW, D, outB_, oF, oB, D, Mq, D, D, ACT_NONE);
  ws.off = mark;
}

extern "C" void kernel_launch(void* const* d_in, const int* in_sizes, int n_in,
                              void* d_out, int out_size, void* d_ws, size_t ws_size,
                              hipStream_t stream){
  (void)in_sizes; (void)n_in; (void)out_size;
  Wspace ws{(char*)d_ws, 0, ws_size};
  const float* x = (const float*)d_in[0];
  auto F = [&](int i){ return (const float*)d_in[i]; };
  auto cvt = [&](const float* s, long n) -> const bf16* {
    bf16* d = (bf16*)walloc(ws, (size_t)n * 2);
    k_cvt<<<(n + 255) / 256, 256, 0, stream>>>(s, d, n);
    return d;
  };
  auto cvtT = [&](const float* s, int K, int N) -> const bf16* {  // s[K,N] -> [N,K]
    bf16* d = (bf16*)walloc(ws, (size_t)K * N * 2);
    dim3 g((N + 15) / 16, (K + 15) / 16);
    k_cvtT<<<g, dim3(16, 16), 0, stream>>>(s, d, K, N);
    return d;
  };

  // ---- convert weights (dict order: x, 6 experts x 26, attn, taus, imp, gate)
  EncW ew[6];
  for (int e = 0; e < 6; ++e) {
    const Cfg& c = CFGS[e];
    int base = 1 + e * 26;
    auto mk = [&](int b) -> MambaW {
      MambaW m;
      m.inproj  = cvt(F(b + 0), (long)2 * c.di * 1024);
      m.convw   = F(b + 1);
      m.convb   = F(b + 2);
      m.xproj   = cvt(F(b + 3), (long)(64 + 2 * c.ds) * c.di);
      m.dtproj  = cvt(F(b + 4), (long)c.di * 64);
      m.dtb     = F(b + 5);
      m.Alog    = F(b + 6);
      m.Dv      = F(b + 7);
      m.outproj = cvt(F(b + 8), (long)1024 * c.di);
      return m;
    };
    ew[e].f = mk(base); ew[e].r = mk(base + 9);
    ew[e].ln1g = F(base + 18); ew[e].ln1b = F(base + 19);
    ew[e].w1t = cvtT(F(base + 20), 1024, c.dff); ew[e].b1 = F(base + 21);
    ew[e].w2t = cvtT(F(base + 22), c.dff, 1024); ew[e].b2 = F(base + 23);
    ew[e].ln2g = F(base + 24); ew[e].ln2b = F(base + 25);
  }
  const bf16* modInW  = cvt(F(157), (long)3 * 1024 * 1024);
  const float* modInB = F(158);
  const bf16* modOutW = cvt(F(159), (long)1024 * 1024);
  const float* modOutB = F(160);
  const bf16* redInW  = cvt(F(161), (long)3 * 1024 * 1024);
  const float* redInB = F(162);
  const bf16* redOutW = cvt(F(163), (long)1024 * 1024);
  const float* redOutB = F(164);
  const float* tauM = F(165); const float* tauR = F(166); const float* tauG = F(167);
  const bf16* impW1t = cvtT(F(168), 1024, 512);
  const float* impB1 = F(169); const float* impLnG = F(170); const float* impLnB = F(171);
  const float* impW2 = F(172); const float* impB2 = F(173);
  const bf16* gw1t = cvtT(F(174), 4096, 1024); const float* gb1 = F(175);
  const bf16* gw2t = cvtT(F(176), 1024, 3);    const float* gb2 = F(177);

  // ---- persistent activation buffers
  const size_t TOK = (size_t)32 * 39 * 1024;
  float* tokens  = (float*)walloc(ws, TOK * 4);
  bf16*  tokensB = (bf16*) walloc(ws, TOK * 2);
  float* synOut  = (float*)walloc(ws, TOK * 4);
  float* redM    = (float*)walloc(ws, TOK * 4);
  float* redOut  = (float*)walloc(ws, TOK * 4);
  float* feats   = (float*)walloc(ws, (size_t)32 * 4 * 1024 * 4);
  bf16*  featsB  = (bf16*) walloc(ws, (size_t)32 * 4 * 1024 * 2);
  bf16*  kmB     = (bf16*) walloc(ws, (size_t)128 * 1024 * 2);
  bf16*  qmB     = (bf16*) walloc(ws, (size_t)128 * 1024 * 2);
  bf16*  tokL2   = (bf16*) walloc(ws, (size_t)1248 * 1024 * 2);
  float* attM    = (float*)walloc(ws, (size_t)128 * 1024 * 4);
  bf16*  attMB   = (bf16*) walloc(ws, (size_t)128 * 1024 * 2);
  float* tImp    = (float*)walloc(ws, (size_t)128 * 512 * 4);
  float* hImp    = (float*)walloc(ws, (size_t)128 * 512 * 4);
  float* logitsI = (float*)walloc(ws, 128 * 4);
  float* modW    = (float*)walloc(ws, 32 * 4 * 4);
  bf16*  g1B     = (bf16*) walloc(ws, (size_t)32 * 1024 * 2);
  float* logits3 = (float*)walloc(ws, 32 * 3 * 4);
  float* interW  = (float*)walloc(ws, 32 * 3 * 4);

  // ---- 4 modality experts write their encoder outputs into token slots
  for (int i = 0; i < 4; ++i)
    run_encoder(stream, ws, x, 39, RSTART[i], RLEN[i], CFGS[i], ew[i],
                tokens, tokensB, 39, RSTART[i]);
  // per-modality mean features
  for (int i = 0; i < 4; ++i)
    k_mean<<<(32 * 1024) / 256, 256, 0, stream>>>(tokens + (size_t)RSTART[i] * 1024,
                                                  39, RLEN[i], feats, featsB, i);
  // synergy expert over concatenated tokens
  run_encoder(stream, ws, tokens, 39, 0, 39, CFGS[4], ew[4], synOut, nullptr, 39, 0);

  // normalized q/k inputs
  k_l2scale<<<128, 256, 0, stream>>>(feats, kmB, 1024, nullptr);
  k_l2scale<<<128, 256, 0, stream>>>(feats, qmB, 1024, tauM);
  k_l2scale<<<1248, 256, 0, stream>>>(tokens, tokL2, 1024, tauR);

  // modality-importance attention, redundancy attention
  run_mha(stream, ws, qmB, 128, 4, kmB, featsB, modInW, modInB, modOutW, modOutB,
          attM, attMB);
  run_mha(stream, ws, tokL2, 1248, 39, kmB, featsB, redInW, redInB, redOutW, redOutB,
          redM, nullptr);
  // redundancy expert on attended tokens
  run_encoder(stream, ws, redM, 39, 0, 39, CFGS[5], ew[5], redOut, nullptr, 39, 0);

  // importance MLP -> mod_w
  gemm(stream, attMB, 1024, impW1t, 1024, impB1, tImp, nullptr, 512,
       128, 512, 1024, ACT_NONE);
  k_ln<<<128, 256, 0, stream>>>(tImp, nullptr, nullptr, 0, impLnG, impLnB,
                                4, 512, 1, hImp, nullptr, 4, 0);
  k_rowdot<<<16, 256, 0, stream>>>(hImp, impW2, impB2, logitsI, 128, 512);
  k_softmax<<<1, 64, 0, stream>>>(logitsI, modW, 32, 4, nullptr);

  // interaction gate -> inter_w
  gemm(stream, featsB, 4096, gw1t, 4096, gb1, nullptr, g1B, 1024,
       32, 1024, 4096, ACT_RELU);
  gemm(stream, g1B, 1024, gw2t, 1024, gb2, logits3, nullptr, 3,
       32, 3, 1024, ACT_NONE);
  k_softmax<<<1, 64, 0, stream>>>(logits3, interW, 32, 3, tauG);

  // gated combination + combined weights output
  float* out  = (float*)d_out;
  float* comb = out + TOK;
  k_combine<<<((long)TOK + 255) / 256, 256, 0, stream>>>(tokens, synOut, redOut,
                                                         modW, interW, out, comb);
}
